// PointNetPlusPlus_14963666059795
// MI455X (gfx1250) — matrix operand order repaired and checked by reading, not compiled, not run on hardware
//
#include <hip/hip_runtime.h>
#include <hip/hip_bf16.h>

// ---------------------------------------------------------------------------
// PointNet++ forward for MI455X (gfx1250, wave32).
// All MLP GEMMs run on v_wmma_f32_16x16x32_bf16 (bf16 in, f32 accum) with a
// fused bias + batchnorm-scale + relu epilogue. Each wave computes a 16x64
// output tile (4 accumulators, A fragment reused by 4 WMMAs). The k-step
// issues all 10 b128 loads before the 4 WMMAs so waits are partial
// (loadcnt<=6/4/2/0) and the scheduler can pipeline across iterations.
// Ball-query uses wave32 ballot compaction. FPS uses a block-per-batch LDS
// argmax reduction.
// ---------------------------------------------------------------------------

typedef __attribute__((ext_vector_type(8)))  float          v8f;
typedef __attribute__((ext_vector_type(8)))  unsigned short v8u;
typedef __attribute__((ext_vector_type(16))) unsigned short v16u;
typedef __attribute__((ext_vector_type(16))) __bf16         v16bf;

#define BB   16
#define NPTS 4096

static __device__ __forceinline__ unsigned short f2bf(float f) {
  unsigned u = __builtin_bit_cast(unsigned, f);
  u += 0x7FFFu + ((u >> 16) & 1u);                 // round-to-nearest-even
  return (unsigned short)(u >> 16);
}
static __device__ __forceinline__ float bf2f(unsigned short h) {
  unsigned u = ((unsigned)h) << 16;
  return __builtin_bit_cast(float, u);
}
static __device__ __forceinline__ v16bf ldfrag(const unsigned short* p0,
                                               const unsigned short* p1) {
  v8u lo = *(const v8u*)p0;
  v8u hi = *(const v8u*)p1;
  return __builtin_bit_cast(
      v16bf, __builtin_shufflevector(lo, hi, 0,1,2,3,4,5,6,7,8,9,10,11,12,13,14,15));
}

// --------------------------- weight packing (f32 -> bf16, Cin padded) -------
__global__ void k_pack_w(const float* __restrict__ W, unsigned short* __restrict__ o,
                         int cin, int cinp, int total) {
  int i = blockIdx.x * 256 + threadIdx.x;
  if (i >= total) return;
  int c = i % cinp, r = i / cinp;
  o[i] = (c < cin) ? f2bf(W[(size_t)r * cin + c]) : (unsigned short)0;
}

// --------------------------- farthest point sampling ------------------------
__global__ void k_fps(const float* __restrict__ xyz, int* __restrict__ cent,
                      float* __restrict__ dist, int N, int S) {
  int b = blockIdx.x;
  const float* base = xyz + (size_t)b * N * 3;
  float* d = dist + (size_t)b * N;
  int* out = cent + (size_t)b * S;
  int tid = threadIdx.x;
  for (int i = tid; i < N; i += 256) d[i] = 1e10f;
  __shared__ float sv[256];
  __shared__ int   si[256];
  __syncthreads();
  int far = 0;
  for (int s = 0; s < S; ++s) {
    if (tid == 0) out[s] = far;
    float cx = base[far * 3 + 0], cy = base[far * 3 + 1], cz = base[far * 3 + 2];
    float bestv = -1.0f; int besti = 0x7fffffff;
    for (int i = tid; i < N; i += 256) {
      float dx = base[i * 3 + 0] - cx, dy = base[i * 3 + 1] - cy, dz = base[i * 3 + 2] - cz;
      float nd = dx * dx + dy * dy + dz * dz;
      float od = d[i];
      nd = nd < od ? nd : od;
      d[i] = nd;
      if (nd > bestv || (nd == bestv && i < besti)) { bestv = nd; besti = i; }
    }
    sv[tid] = bestv; si[tid] = besti;
    __syncthreads();
    for (int off = 128; off; off >>= 1) {
      if (tid < off) {
        if (sv[tid + off] > sv[tid] ||
            (sv[tid + off] == sv[tid] && si[tid + off] < si[tid])) {
          sv[tid] = sv[tid + off]; si[tid] = si[tid + off];
        }
      }
      __syncthreads();
    }
    far = si[0];
    __syncthreads();
  }
}

// --------------------------- gather centroid coords -------------------------
__global__ void k_gather_xyz(const float* __restrict__ xyz, const int* __restrict__ cent,
                             float* __restrict__ nx, int N, int S) {
  int i = blockIdx.x * 256 + threadIdx.x;
  if (i >= BB * S) return;
  int b = i / S;
  int g = cent[i];
  const float* p = xyz + ((size_t)b * N + g) * 3;
  nx[i * 3 + 0] = p[0]; nx[i * 3 + 1] = p[1]; nx[i * 3 + 2] = p[2];
}

// --------------------------- ball query (wave32 ballot compaction) ----------
__global__ void k_ball(const float* __restrict__ xyz, const float* __restrict__ nx,
                       int* __restrict__ gi, int N, int S, int ns, float r2) {
  int wave = (int)((blockIdx.x * 256 + threadIdx.x) >> 5);
  int lane = threadIdx.x & 31;
  if (wave >= BB * S) return;
  int b = wave / S;
  const float* base = xyz + (size_t)b * N * 3;
  float cx = nx[wave * 3 + 0], cy = nx[wave * 3 + 1], cz = nx[wave * 3 + 2];
  int* out = gi + (size_t)wave * ns;
  int have = 0, first = 0;
  for (int c0 = 0; c0 < N && have < ns; c0 += 32) {
    int j = c0 + lane;
    float dx = base[j * 3 + 0] - cx, dy = base[j * 3 + 1] - cy, dz = base[j * 3 + 2] - cz;
    float dd = dx * dx + dy * dy + dz * dz;
    bool pred = (dd <= r2);
    unsigned mask = __builtin_amdgcn_ballot_w32(pred);
    if (mask) {
      if (have == 0) first = c0 + __builtin_ctz(mask);
      int pos = have + __builtin_popcount(mask & ((1u << lane) - 1u));
      if (pred && pos < ns) out[pos] = j;
      have += __builtin_popcount(mask);
    }
  }
  if (have > ns) have = ns;
  for (int k = have + lane; k < ns; k += 32) out[k] = first;
}

// --------------------------- grouped feature build (rel-xyz ++ feats) -------
__global__ void k_group(const float* __restrict__ xyz, const float* __restrict__ nx,
                        const float* __restrict__ pts, const int* __restrict__ gi,
                        unsigned short* __restrict__ X,
                        int N, int S, int ns, int C, int cinp, size_t total) {
  size_t i = (size_t)blockIdx.x * 256 + threadIdx.x;
  if (i >= total) return;
  int c = (int)(i % cinp);
  size_t row = i / cinp;
  int j = (int)(row % ns);
  size_t bs = row / ns;          // b*S + s
  int b = (int)(bs / S);
  int g = gi[bs * ns + j];
  float v = 0.0f;
  if (c < 3)          v = xyz[((size_t)b * N + g) * 3 + c] - nx[bs * 3 + c];
  else if (c < 3 + C) v = pts[((size_t)b * N + g) * C + (c - 3)];
  X[i] = f2bf(v);
}

// --------------------------- sa3 group-all feature build --------------------
__global__ void k_feats_sa3(const float* __restrict__ x2, const float* __restrict__ p2,
                            unsigned short* __restrict__ X, int C, int cinp, size_t total) {
  size_t i = (size_t)blockIdx.x * 256 + threadIdx.x;
  if (i >= total) return;
  int c = (int)(i % cinp);
  size_t row = i / cinp;         // b*S + s
  float v = 0.0f;
  if (c < 3)          v = x2[row * 3 + c];
  else if (c < 3 + C) v = p2[row * C + (c - 3)];
  X[i] = f2bf(v);
}

// --------------------------- fp3 feature build (concat + broadcast) ---------
__global__ void k_feats_fp3(const float* __restrict__ p, const float* __restrict__ gl,
                            unsigned short* __restrict__ X, int S, int C1, int C2,
                            size_t total) {
  size_t i = (size_t)blockIdx.x * 256 + threadIdx.x;
  if (i >= total) return;
  int cols = C1 + C2;
  int c = (int)(i % cols);
  size_t row = i / cols;
  int b = (int)(row / S);
  float v = (c < C1) ? p[row * C1 + c] : gl[(size_t)b * C2 + (c - C1)];
  X[i] = f2bf(v);
}

// --------------------------- 3-NN search ------------------------------------
__global__ void k_knn3(const float* __restrict__ x1, const float* __restrict__ x2,
                       int n, int s, int* __restrict__ ki, float* __restrict__ kw) {
  int i = blockIdx.x * 256 + threadIdx.x;
  if (i >= BB * n) return;
  int b = i / n;
  float ax = x1[i * 3 + 0], ay = x1[i * 3 + 1], az = x1[i * 3 + 2];
  float d0 = 3.4e38f, d1 = 3.4e38f, d2 = 3.4e38f;
  int i0 = 0, i1 = 0, i2 = 0;
  const float* base = x2 + (size_t)b * s * 3;
  for (int j = 0; j < s; ++j) {
    float dx = base[j * 3 + 0] - ax, dy = base[j * 3 + 1] - ay, dz = base[j * 3 + 2] - az;
    float dd = dx * dx + dy * dy + dz * dz;
    if (dd < d0)      { d2 = d1; i2 = i1; d1 = d0; i1 = i0; d0 = dd; i0 = j; }
    else if (dd < d1) { d2 = d1; i2 = i1; d1 = dd; i1 = j; }
    else if (dd < d2) { d2 = dd; i2 = j; }
  }
  float r0 = 1.0f / (d0 + 1e-8f), r1 = 1.0f / (d1 + 1e-8f), r2 = 1.0f / (d2 + 1e-8f);
  float inv = 1.0f / (r0 + r1 + r2);
  ki[i * 3 + 0] = i0; ki[i * 3 + 1] = i1; ki[i * 3 + 2] = i2;
  kw[i * 3 + 0] = r0 * inv; kw[i * 3 + 1] = r1 * inv; kw[i * 3 + 2] = r2 * inv;
}

// --------------------------- fp feature build (skip ++ 3NN interp) ----------
__global__ void k_feats_interp(const float* __restrict__ p1, int C1,
                               const float* __restrict__ p2, int C2,
                               const int* __restrict__ ki, const float* __restrict__ kw,
                               unsigned short* __restrict__ X, int n, int s, size_t total) {
  size_t i = (size_t)blockIdx.x * 256 + threadIdx.x;
  if (i >= total) return;
  int cols = C1 + C2;
  int c = (int)(i % cols);
  size_t row = i / cols;         // b*n + i
  int b = (int)(row / n);
  float v;
  if (c < C1) {
    v = p1[row * C1 + c];
  } else {
    int cc = c - C1;
    v = 0.0f;
    #pragma unroll
    for (int k = 0; k < 3; ++k)
      v += kw[row * 3 + k] * p2[((size_t)b * s + ki[row * 3 + k]) * C2 + cc];
  }
  X[i] = f2bf(v);
}

// --------------------------- max pool over group dim ------------------------
__global__ void k_maxpool(const unsigned short* __restrict__ Y, float* __restrict__ o,
                          int ns, int C, int total) {
  int i = blockIdx.x * 256 + threadIdx.x;
  if (i >= total) return;
  int c = i % C;
  size_t g = (size_t)(i / C);
  float m = -3.4e38f;
  for (int j = 0; j < ns; ++j) {
    float v = bf2f(Y[((size_t)g * ns + j) * C + c]);
    m = v > m ? v : m;
  }
  o[i] = m;
}

// --------------------------- WMMA GEMM + bias + BN + relu -------------------
// One wave computes a 16(M) x 64(N) output tile with 4 f32 accumulators:
//   Y = relu(gamma*((X*W^T + b)*inv) + beta)
// X: P x Cin (bf16, Cin % 32 == 0, P % 16 == 0)
// W: Cout x Cin (bf16, Cin padded, Cout % 64 == 0)
// All 10 fragment loads of a k-step are issued before its 4 WMMAs, so the
// waits are partial and the scheduler can pipeline across iterations.
__global__ void k_gemm(const unsigned short* __restrict__ X,
                       const unsigned short* __restrict__ Wm,
                       const float* __restrict__ bias,
                       const float* __restrict__ gamma,
                       const float* __restrict__ beta,
                       unsigned short* __restrict__ Ybf,
                       float* __restrict__ Yf,
                       int P, int Cin, int Cout) {
  const int wave = threadIdx.x >> 5;
  const int lane = threadIdx.x & 31;
  const int tilesN = Cout >> 6;              // 64-wide N tiles
  const int tiles = (P >> 4) * tilesN;
  int tile = blockIdx.x * (blockDim.x >> 5) + wave;
  if (tile >= tiles) return;                 // uniform per wave: EXEC stays all-1
  int mt = tile / tilesN, nt = tile - mt * tilesN;
  int m0 = mt << 4, n0 = nt << 6;
  int h = lane >> 4, l = lane & 15;
  const unsigned short* xrow = X + (size_t)(m0 + l) * Cin;
  const unsigned short* wr0 = Wm + (size_t)(n0 + l) * Cin;
  const unsigned short* wr1 = wr0 + (size_t)16 * Cin;
  const unsigned short* wr2 = wr0 + (size_t)32 * Cin;
  const unsigned short* wr3 = wr0 + (size_t)48 * Cin;
  const int ao = 16 * h;                     // A: elem e -> K=16h+e
  const int bo = 16 * h;                     // B: elem e -> K=16h+e (B[k][n] = W[n][k])

  v8f acc0 = {0.f,0.f,0.f,0.f,0.f,0.f,0.f,0.f};
  v8f acc1 = acc0, acc2 = acc0, acc3 = acc0;

  for (int k0 = 0; k0 < Cin; k0 += 32) {
    // issue all fragment loads for this k-step first (A, then B0..B3)
    v16bf a  = ldfrag(xrow + k0 + ao,  xrow + k0 + ao + 8);
    v16bf b0 = ldfrag(wr0 + k0 + bo,   wr0 + k0 + bo + 8);
    v16bf b1 = ldfrag(wr1 + k0 + bo,   wr1 + k0 + bo + 8);
    v16bf b2 = ldfrag(wr2 + k0 + bo,   wr2 + k0 + bo + 8);
    v16bf b3 = ldfrag(wr3 + k0 + bo,   wr3 + k0 + bo + 8);
    acc0 = __builtin_amdgcn_wmma_f32_16x16x32_bf16(false, a, false, b0, (short)0, acc0, false, false);
    acc1 = __builtin_amdgcn_wmma_f32_16x16x32_bf16(false, a, false, b1, (short)0, acc1, false, false);
    acc2 = __builtin_amdgcn_wmma_f32_16x16x32_bf16(false, a, false, b2, (short)0, acc2, false, false);
    acc3 = __builtin_amdgcn_wmma_f32_16x16x32_bf16(false, a, false, b3, (short)0, acc3, false, false);
  }

  const float inv = 0.9999950000374997f;     // 1/sqrt(1 + 1e-5)
  v8f* accs[4] = {&acc0, &acc1, &acc2, &acc3};
  #pragma unroll
  for (int t = 0; t < 4; ++t) {
    int n = n0 + t * 16 + l;
    float bi = bias[n], ga = gamma[n], be = beta[n];
    #pragma unroll
    for (int r = 0; r < 8; ++r) {
      int m = m0 + r + 8 * h;                // D layout: VGPR r -> M = r + 8*half
      float v = (*accs[t])[r] + bi;
      v = ga * (v * inv) + be;
      v = v > 0.f ? v : 0.f;
      size_t o = (size_t)m * Cout + n;
      if (Ybf) Ybf[o] = f2bf(v);
      if (Yf)  Yf[o]  = v;
    }
  }
}

// ===========================================================================
extern "C" void kernel_launch(void* const* d_in, const int* in_sizes, int n_in,
                              void* d_out, int out_size, void* d_ws, size_t ws_size,
                              hipStream_t stream) {
  // Layer tables in pipeline order: sa1(3), sa2(3), sa3(3), fp3(2), fp2(2), fp1(3)
  static const int cin[16]  = {6,64,64, 131,128,128, 259,256,512, 1280,256, 384,256, 128,128,128};
  static const int cout[16] = {64,64,128, 128,128,256, 256,512,1024, 256,256, 256,128, 128,128,128};
  int cinp[16];
  for (int i = 0; i < 16; ++i) cinp[i] = (cin[i] + 31) & ~31;

  static const int glen[6]   = {3,3,3,2,2,3};
  static const int gstart[6] = {0,3,6,9,11,13};

  const float *Wf[16], *Bf[16], *Gf[16], *Bt[16];
  const float* xyz;
  bool insertion = (in_sizes[0] == BB * NPTS * 3);
  if (insertion) {
    // {xyz, idx, params{sa1,sa2,sa3,fp3,fp2,fp1; each layer W,b,gamma,beta}}
    xyz = (const float*)d_in[0];
    int p = 2;
    for (int g = 0; g < 6; ++g)
      for (int l = 0; l < glen[g]; ++l) {
        int li = gstart[g] + l;
        Wf[li] = (const float*)d_in[p++]; Bf[li] = (const float*)d_in[p++];
        Gf[li] = (const float*)d_in[p++]; Bt[li] = (const float*)d_in[p++];
      }
  } else {
    // jax tree-flatten (alphabetical): idx, params{fp1,fp2,fp3,sa1,sa2,sa3; W,b,beta,gamma}, xyz
    static const int aorder[6] = {5,4,3,0,1,2};
    int p = 1;
    for (int a = 0; a < 6; ++a) {
      int g = aorder[a];
      for (int l = 0; l < glen[g]; ++l) {
        int li = gstart[g] + l;
        Wf[li] = (const float*)d_in[p++]; Bf[li] = (const float*)d_in[p++];
        Bt[li] = (const float*)d_in[p++]; Gf[li] = (const float*)d_in[p++];
      }
    }
    xyz = (const float*)d_in[p];
  }

  // ---- workspace bump allocator ----
  char* wp = (char*)d_ws;
  auto alloc = [&](size_t bytes) -> void* {
    void* r = (void*)wp;
    wp += (bytes + 255) & ~(size_t)255;
    return r;
  };
  unsigned short* pingA = (unsigned short*)alloc((size_t)262144 * 128 * 2);
  unsigned short* pingB = (unsigned short*)alloc((size_t)262144 * 128 * 2);
  unsigned short* Wbf[16];
  for (int i = 0; i < 16; ++i) Wbf[i] = (unsigned short*)alloc((size_t)cout[i] * cinp[i] * 2);
  float* dist  = (float*)alloc((size_t)BB * NPTS * 4);
  int*   cent1 = (int*)  alloc((size_t)BB * 512 * 4);
  int*   cent2 = (int*)  alloc((size_t)BB * 128 * 4);
  float* l1x   = (float*)alloc((size_t)BB * 512 * 3 * 4);
  float* l2x   = (float*)alloc((size_t)BB * 128 * 3 * 4);
  int*   gi1   = (int*)  alloc((size_t)BB * 512 * 32 * 4);
  int*   gi2   = (int*)  alloc((size_t)BB * 128 * 64 * 4);
  float* l1p   = (float*)alloc((size_t)BB * 512 * 128 * 4);
  float* l2p   = (float*)alloc((size_t)BB * 128 * 256 * 4);
  float* l3p   = (float*)alloc((size_t)BB * 1024 * 4);
  float* l2p2  = (float*)alloc((size_t)BB * 128 * 256 * 4);
  float* l1p2  = (float*)alloc((size_t)BB * 512 * 128 * 4);
  int*   ki2   = (int*)  alloc((size_t)BB * 512 * 3 * 4);
  float* kw2   = (float*)alloc((size_t)BB * 512 * 3 * 4);
  int*   ki1   = (int*)  alloc((size_t)BB * NPTS * 3 * 4);
  float* kw1   = (float*)alloc((size_t)BB * NPTS * 3 * 4);

  auto blocks = [](size_t total) { return (unsigned)((total + 255) / 256); };
  auto gemm = [&](const unsigned short* Xp, int li, unsigned short* Yb, float* Yf, int P) {
    int tiles = (P / 16) * (cout[li] / 64);
    k_gemm<<<(tiles + 7) / 8, 256, 0, stream>>>(Xp, Wbf[li], Bf[li], Gf[li], Bt[li],
                                                Yb, Yf, P, cinp[li], cout[li]);
  };

  // ---- pack all weights to padded bf16 ----
  for (int i = 0; i < 16; ++i) {
    int tot = cout[i] * cinp[i];
    k_pack_w<<<blocks(tot), 256, 0, stream>>>(Wf[i], Wbf[i], cin[i], cinp[i], tot);
  }

  // ---- SA1: N=4096 -> 512 centroids, ns=32, r=0.2 ----
  k_fps<<<BB, 256, 0, stream>>>(xyz, cent1, dist, NPTS, 512);
  k_gather_xyz<<<blocks(BB * 512), 256, 0, stream>>>(xyz, cent1, l1x, NPTS, 512);
  k_ball<<<blocks((size_t)BB * 512 * 32), 256, 0, stream>>>(xyz, l1x, gi1, NPTS, 512, 32, 0.04f);
  {
    size_t tot = (size_t)BB * 512 * 32 * cinp[0];
    k_group<<<blocks(tot), 256, 0, stream>>>(xyz, l1x, xyz, gi1, pingA, NPTS, 512, 32, 3, cinp[0], tot);
  }
  gemm(pingA, 0, pingB, nullptr, BB * 512 * 32);
  gemm(pingB, 1, pingA, nullptr, BB * 512 * 32);
  gemm(pingA, 2, pingB, nullptr, BB * 512 * 32);
  k_maxpool<<<blocks((size_t)BB * 512 * 128), 256, 0, stream>>>(pingB, l1p, 32, 128, BB * 512 * 128);

  // ---- SA2: 512 -> 128 centroids, ns=64, r=0.4 ----
  k_fps<<<BB, 256, 0, stream>>>(l1x, cent2, dist, 512, 128);
  k_gather_xyz<<<blocks(BB * 128), 256, 0, stream>>>(l1x, cent2, l2x, 512, 128);
  k_ball<<<blocks((size_t)BB * 128 * 32), 256, 0, stream>>>(l1x, l2x, gi2, 512, 128, 64, 0.16f);
  {
    size_t tot = (size_t)BB * 128 * 64 * cinp[3];
    k_group<<<blocks(tot), 256, 0, stream>>>(l1x, l2x, l1p, gi2, pingA, 512, 128, 64, 128, cinp[3], tot);
  }
  gemm(pingA, 3, pingB, nullptr, BB * 128 * 64);
  gemm(pingB, 4, pingA, nullptr, BB * 128 * 64);
  gemm(pingA, 5, pingB, nullptr, BB * 128 * 64);
  k_maxpool<<<blocks((size_t)BB * 128 * 256), 256, 0, stream>>>(pingB, l2p, 64, 256, BB * 128 * 256);

  // ---- SA3: group-all over 128 points ----
  {
    size_t tot = (size_t)BB * 128 * cinp[6];
    k_feats_sa3<<<blocks(tot), 256, 0, stream>>>(l2x, l2p, pingA, 256, cinp[6], tot);
  }
  gemm(pingA, 6, pingB, nullptr, BB * 128);
  gemm(pingB, 7, pingA, nullptr, BB * 128);
  gemm(pingA, 8, pingB, nullptr, BB * 128);
  k_maxpool<<<blocks((size_t)BB * 1024), 256, 0, stream>>>(pingB, l3p, 128, 1024, BB * 1024);

  // ---- FP3: broadcast global feature, MLP [1280,256,256] ----
  {
    size_t tot = (size_t)BB * 128 * 1280;
    k_feats_fp3<<<blocks(tot), 256, 0, stream>>>(l2p, l3p, pingA, 128, 256, 1024, tot);
  }
  gemm(pingA, 9,  pingB, nullptr, BB * 128);
  gemm(pingB, 10, pingA, l2p2,    BB * 128);

  // ---- FP2: 3-NN interp 128->512, concat l1p, MLP [384,256,128] ----
  k_knn3<<<blocks(BB * 512), 256, 0, stream>>>(l1x, l2x, 512, 128, ki2, kw2);
  {
    size_t tot = (size_t)BB * 512 * 384;
    k_feats_interp<<<blocks(tot), 256, 0, stream>>>(l1p, 128, l2p2, 256, ki2, kw2, pingA, 512, 128, tot);
  }
  gemm(pingA, 11, pingB, nullptr, BB * 512);
  gemm(pingB, 12, pingA, l1p2,    BB * 512);

  // ---- FP1: 3-NN interp 512->4096, MLP [128,128,128,128] -> d_out ----
  k_knn3<<<blocks((size_t)BB * NPTS), 256, 0, stream>>>(xyz, l1x, NPTS, 512, ki1, kw1);
  {
    size_t tot = (size_t)BB * NPTS * 128;
    k_feats_interp<<<blocks(tot), 256, 0, stream>>>(nullptr, 0, l1p2, 128, ki1, kw1, pingA, NPTS, 512, tot);
  }
  gemm(pingA, 13, pingB, nullptr, BB * NPTS);
  gemm(pingB, 14, pingA, nullptr, BB * NPTS);
  gemm(pingA, 15, nullptr, (float*)d_out, BB * NPTS);
}
